// PointNet_69518340653116
// MI455X (gfx1250) — matrix operand order repaired and checked by reading, not compiled, not run on hardware
//
#include <hip/hip_runtime.h>
#include <hip/hip_bf16.h>
#include <math.h>

typedef __attribute__((ext_vector_type(16))) _Float16 v16h;
typedef __attribute__((ext_vector_type(8)))  float    v8f;
typedef __attribute__((ext_vector_type(2)))  _Float16 h2;

#define NPTS  65536
#define DDIM  50
#define ODIM  32
#define PTS_PER_WAVE   16
#define WAVES_PER_BLK  8
#define PTS_PER_BLK    (PTS_PER_WAVE * WAVES_PER_BLK)

// dynamic LDS layout:
//  swz_w1  : 4096 halves   ( 64x 64, 4 n-tiles x 2 k-chunks)  @ 0
//  swz_w2  : 8192 halves   ( 64x128, 8 x 2)                   @ 8192 B
//  swz_rw0 : 8192 halves   (128x 64, 4 x 4)                   @ 24576 B
//  swz_rw1 : 4096 halves   ( 64x 64, 4 x 2)                   @ 40960 B
//  swz_rw2 : 4096 halves   ( 64x 64, 4 x 2)                   @ 49152 B
//  sb      : 320 floats    (h_b2[128], r_b0[64], r_b1[64], r_b2[64]) @ 57344 B
//  mbits   : 128 u64       (per-point packed mask bitmasks)   @ 58624 B
//  scratch : 8 * 2048 floats (per-wave transpose scratch)     @ 59648 B
#define LDS_BYTES (57344 + 1280 + 1024 + 8 * 2048 * 4)

__device__ __forceinline__ v8f wmma16(v16h a, v16h b, v8f c) {
    return __builtin_amdgcn_wmma_f32_16x16x32_f16(false, a, false, b, (short)0, c,
                                                  false, false);
}

// K index served by element j (0..15) of the f16 A operand, chunk c of 32,
// per the CDNA5 16-bit 16x32 A layout (lanes 0-15: K 0-7,16-23; lanes 16-31: +8).
__device__ __forceinline__ int kidxA(int lane, int c, int j) {
    return (c << 5) + ((lane >> 4) << 3) + (j < 8 ? j : j + 8);
}

// Pre-swizzle a row-major KxN f32 weight into per-lane-contiguous f16 B-operand
// order: dst[((t*NC+c)*32 + lane)*16 + j] = W[c*32 + (lane>=16?8:0) + (j<8?j:j+8)][t*16 + lane%16]
__device__ void swizzleB(const float* __restrict__ W, _Float16* dst, int K, int Nn) {
    const int NT = Nn >> 4, NC = K >> 5;
    const int total = NT * NC * 32;
    for (int i = threadIdx.x; i < total; i += blockDim.x) {
        const int lane = i & 31;
        const int tc   = i >> 5;
        const int c    = tc % NC;
        const int t    = tc / NC;
        const int n    = (t << 4) + (lane & 15);
        const int k0   = (c << 5) + ((lane >> 4) << 3);
        _Float16* o = dst + i * 16;
        #pragma unroll
        for (int j = 0; j < 16; ++j)
            o[j] = (_Float16)W[(k0 + (j < 8 ? j : j + 8)) * Nn + n];
    }
}

__global__ __launch_bounds__(256) void pointnet_fused(
    const float* __restrict__ x, const unsigned char* __restrict__ mask,
    const float* __restrict__ h_w0, const float* __restrict__ h_b0,
    const float* __restrict__ h_w1, const float* __restrict__ h_b1,
    const float* __restrict__ h_w2, const float* __restrict__ h_b2,
    const float* __restrict__ r_w0, const float* __restrict__ r_b0,
    const float* __restrict__ r_w1, const float* __restrict__ r_b1,
    const float* __restrict__ r_w2, const float* __restrict__ r_b2,
    float* __restrict__ out)
{
    extern __shared__ __align__(16) unsigned char smem_raw[];
    _Float16* swz_w1  = (_Float16*)smem_raw;
    _Float16* swz_w2  = swz_w1  + 4096;
    _Float16* swz_rw0 = swz_w2  + 8192;
    _Float16* swz_rw1 = swz_rw0 + 8192;
    _Float16* swz_rw2 = swz_rw1 + 4096;
    float*    sb      = (float*)(swz_rw2 + 4096);
    unsigned long long* mbits = (unsigned long long*)(sb + 320);
    float*    scratch = (float*)(mbits + 128);

    swizzleB(h_w1, swz_w1,  64,  64);
    swizzleB(h_w2, swz_w2,  64, 128);
    swizzleB(r_w0, swz_rw0, 128, 64);
    swizzleB(r_w1, swz_rw1, 64,  64);
    swizzleB(r_w2, swz_rw2, 64,  64);
    for (int i = threadIdx.x; i < 320; i += blockDim.x) {
        float v;
        if      (i < 128) v = h_b2[i];
        else if (i < 192) v = r_b0[i - 128];
        else if (i < 256) v = r_b1[i - 192];
        else              v = r_b2[i - 256];
        sb[i] = v;
    }
    // pack each point's 50 mask bytes into one u64 bitmask
    if (threadIdx.x < PTS_PER_BLK) {
        const unsigned char* mp = mask + (blockIdx.x * PTS_PER_BLK + threadIdx.x) * DDIM;
        unsigned long long bits = 0ull;
        #pragma unroll
        for (int d = 0; d < DDIM; ++d)
            bits |= (unsigned long long)(mp[d] ? 1u : 0u) << d;
        mbits[threadIdx.x] = bits;
    }
    __syncthreads();

    const int lane  = threadIdx.x & 31;
    const int wv    = threadIdx.x >> 5;
    const int m     = lane & 15;      // A-row / C-column index
    const int hb    = lane >> 4;      // lane-half
    const int mr    = hb << 3;        // C-layout row base
    const int n0    = blockIdx.x * PTS_PER_BLK + wv * PTS_PER_WAVE;
    const int point = n0 + m;
    float* sc = scratch + wv * 2048;

    // hidden layer-1 weights resident in registers, B-operand order
    v16h w1b[8];
    #pragma unroll
    for (int i = 0; i < 8; ++i)
        w1b[i] = *(const v16h*)(swz_w1 + (i * 32 + lane) * 16);

    // layer-0 weights (2 -> 64) as packed-f16 pairs, already in A lane order
    h2 w0xp[16], w0dp[16], b0p[16];
    #pragma unroll
    for (int i = 0; i < 16; ++i) {
        const int c  = i >> 3, p = i & 7;
        const int ka = kidxA(lane, c, 2 * p);
        const int kb = kidxA(lane, c, 2 * p + 1);
        h2 t0 = { (_Float16)h_w0[ka],      (_Float16)h_w0[kb]      }; w0xp[i] = t0;
        h2 t1 = { (_Float16)h_w0[64 + ka], (_Float16)h_w0[64 + kb] }; w0dp[i] = t1;
        h2 t2 = { (_Float16)h_b0[ka],      (_Float16)h_b0[kb]      }; b0p[i]  = t2;
    }
    float b1r[4];
    #pragma unroll
    for (int t = 0; t < 4; ++t) b1r[t] = h_b1[t * 16 + m];

    // per-lane mask bitmasks for the 8 C-layout rows this lane owns
    unsigned long long rowbits[8];
    #pragma unroll
    for (int r = 0; r < 8; ++r)
        rowbits[r] = mbits[wv * 16 + mr + r];
    const float cnt = (float)__builtin_popcountll(mbits[wv * 16 + m]);

    // S = sum_d mask * relu(h1_d)   (16x64 f32, C layout)
    v8f S[4];
    #pragma unroll
    for (int t = 0; t < 4; ++t)
        #pragma unroll
        for (int e = 0; e < 8; ++e) S[t][e] = 0.0f;

    const float* xrow = x + point * DDIM;
    const h2 zero2 = { (_Float16)0.0f, (_Float16)0.0f };

    float xnext = xrow[0];
    #pragma unroll 1
    for (int d = 0; d < DDIM; ++d) {
        const float xv = xnext;
        if (d + 1 < DDIM) xnext = xrow[d + 1];
        // mask bits for this d, one per owned row
        float msf[8];
        #pragma unroll
        for (int r = 0; r < 8; ++r) {
            msf[r] = (float)((unsigned)rowbits[r] & 1u);
            rowbits[r] >>= 1;
        }
        // ---- layer 0: [x, d] -> 64, relu, emitted directly in A-operand order
        const _Float16 xh = (_Float16)xv;
        const _Float16 dh = (_Float16)(float)d;
        const h2 x2 = { xh, xh };
        const h2 d2 = { dh, dh };
        v16h a0, a1;
        #pragma unroll
        for (int i = 0; i < 16; ++i) {
            h2 t0 = d2 * w0dp[i] + b0p[i];
            h2 v  = x2 * w0xp[i] + t0;
            v = __builtin_elementwise_max(v, zero2);
            if (i < 8) { a0[2 * i] = v.x;       a0[2 * i + 1] = v.y; }
            else       { a1[2 * (i - 8)] = v.x; a1[2 * (i - 8) + 1] = v.y; }
        }
        // ---- layer 1: 16x64 @ 64x64 + b1 (B resident in registers)
        // ---- then S += mask * relu(h1)  (pooling hoisted past linear layer 2)
        #pragma unroll
        for (int t = 0; t < 4; ++t) {
            v8f cc;
            #pragma unroll
            for (int e = 0; e < 8; ++e) cc[e] = b1r[t];
            cc = wmma16(a0, w1b[t * 2 + 0], cc);
            cc = wmma16(a1, w1b[t * 2 + 1], cc);
            #pragma unroll
            for (int e = 0; e < 8; ++e)
                S[t][e] = fmaf(msf[e], fmaxf(cc[e], 0.0f), S[t][e]);
        }
    }

    // ---- S (C layout) -> LDS -> A operand (transpose), once per wave
    #pragma unroll
    for (int t = 0; t < 4; ++t)
        #pragma unroll
        for (int r = 0; r < 8; ++r)
            sc[(mr + r) * 64 + t * 16 + m] = S[t][r];
    v16h aS[2];
    #pragma unroll
    for (int c = 0; c < 2; ++c)
        #pragma unroll
        for (int j = 0; j < 16; ++j)
            aS[c][j] = (_Float16)sc[m * 64 + kidxA(lane, c, j)];

    // ---- layer 2 (hoisted out of d loop): pooled = S @ w2  (+ cnt*b2 later)
    v8f pooled[8];
    #pragma unroll
    for (int t = 0; t < 8; ++t) {
        v8f cc;
        #pragma unroll
        for (int e = 0; e < 8; ++e) cc[e] = 0.0f;
        v16h bb0 = *(const v16h*)(swz_w2 + ((t * 2 + 0) * 32 + lane) * 16);
        cc = wmma16(aS[0], bb0, cc);
        v16h bb1 = *(const v16h*)(swz_w2 + ((t * 2 + 1) * 32 + lane) * 16);
        cc = wmma16(aS[1], bb1, cc);
        pooled[t] = cc;
    }

    // ---- pooled (16x128) -> LDS, reload as A with cnt*b2 folded in
    #pragma unroll
    for (int t = 0; t < 8; ++t)
        #pragma unroll
        for (int r = 0; r < 8; ++r)
            sc[(mr + r) * 128 + t * 16 + m] = pooled[t][r];
    v16h ar[4];
    #pragma unroll
    for (int c = 0; c < 4; ++c)
        #pragma unroll
        for (int j = 0; j < 16; ++j) {
            const int k = kidxA(lane, c, j);
            ar[c][j] = (_Float16)(sc[m * 128 + k] + cnt * sb[k]);
        }

    // ---- r0 = relu(pooled @ r_w0 + r_b0) : 128 -> 64
    v8f r0v[4];
    #pragma unroll
    for (int t = 0; t < 4; ++t) {
        const float bv = sb[128 + t * 16 + m];
        v8f cc;
        #pragma unroll
        for (int e = 0; e < 8; ++e) cc[e] = bv;
        #pragma unroll
        for (int c = 0; c < 4; ++c) {
            v16h bb = *(const v16h*)(swz_rw0 + ((t * 4 + c) * 32 + lane) * 16);
            cc = wmma16(ar[c], bb, cc);
        }
        #pragma unroll
        for (int e = 0; e < 8; ++e) cc[e] = fmaxf(cc[e], 0.0f);
        r0v[t] = cc;
    }
    #pragma unroll
    for (int t = 0; t < 4; ++t)
        #pragma unroll
        for (int r = 0; r < 8; ++r)
            sc[(mr + r) * 64 + t * 16 + m] = r0v[t][r];
    v16h a1r[2];
    #pragma unroll
    for (int c = 0; c < 2; ++c)
        #pragma unroll
        for (int j = 0; j < 16; ++j)
            a1r[c][j] = (_Float16)sc[m * 64 + kidxA(lane, c, j)];

    // ---- r1 = relu(r0 @ r_w1 + r_b1) : 64 -> 64
    v8f r1v[4];
    #pragma unroll
    for (int t = 0; t < 4; ++t) {
        const float bv = sb[192 + t * 16 + m];
        v8f cc;
        #pragma unroll
        for (int e = 0; e < 8; ++e) cc[e] = bv;
        #pragma unroll
        for (int c = 0; c < 2; ++c) {
            v16h bb = *(const v16h*)(swz_rw1 + ((t * 2 + c) * 32 + lane) * 16);
            cc = wmma16(a1r[c], bb, cc);
        }
        #pragma unroll
        for (int e = 0; e < 8; ++e) cc[e] = fmaxf(cc[e], 0.0f);
        r1v[t] = cc;
    }
    #pragma unroll
    for (int t = 0; t < 4; ++t)
        #pragma unroll
        for (int r = 0; r < 8; ++r)
            sc[(mr + r) * 64 + t * 16 + m] = r1v[t][r];
    v16h a2r[2];
    #pragma unroll
    for (int c = 0; c < 2; ++c)
        #pragma unroll
        for (int j = 0; j < 16; ++j)
            a2r[c][j] = (_Float16)sc[m * 64 + kidxA(lane, c, j)];

    // ---- r2 = r1 @ r_w2 + r_b2 : 64 -> 64; split mu / softplus(sigma)
    #pragma unroll
    for (int t = 0; t < 4; ++t) {
        const float bv = sb[256 + t * 16 + m];
        v8f cc;
        #pragma unroll
        for (int e = 0; e < 8; ++e) cc[e] = bv;
        #pragma unroll
        for (int c = 0; c < 2; ++c) {
            v16h bb = *(const v16h*)(swz_rw2 + ((t * 2 + c) * 32 + lane) * 16);
            cc = wmma16(a2r[c], bb, cc);
        }
        #pragma unroll
        for (int r = 0; r < 8; ++r) {
            const float v  = cc[r];
            const int   pt = n0 + mr + r;
            const int   nn = t * 16 + m;
            if (t < 2) {
                out[pt * ODIM + nn] = v;                          // mu
            } else {
                const float sp = fmaxf(v, 0.0f) + log1pf(expf(-fabsf(v)));
                out[NPTS * ODIM + pt * ODIM + (nn - ODIM)] = sp;  // sigma
            }
        }
    }
}

extern "C" void kernel_launch(void* const* d_in, const int* in_sizes, int n_in,
                              void* d_out, int out_size, void* d_ws, size_t ws_size,
                              hipStream_t stream) {
    (void)in_sizes; (void)n_in; (void)d_ws; (void)ws_size; (void)out_size;
    const float*         x    = (const float*)d_in[0];
    const unsigned char* mask = (const unsigned char*)d_in[1];
    const float* h_w0 = (const float*)d_in[2];
    const float* h_b0 = (const float*)d_in[3];
    const float* h_w1 = (const float*)d_in[4];
    const float* h_b1 = (const float*)d_in[5];
    const float* h_w2 = (const float*)d_in[6];
    const float* h_b2 = (const float*)d_in[7];
    const float* r_w0 = (const float*)d_in[8];
    const float* r_b0 = (const float*)d_in[9];
    const float* r_w1 = (const float*)d_in[10];
    const float* r_b1 = (const float*)d_in[11];
    const float* r_w2 = (const float*)d_in[12];
    const float* r_b2 = (const float*)d_in[13];
    float* out = (float*)d_out;

    dim3 grid(NPTS / PTS_PER_BLK), block(256);
    pointnet_fused<<<grid, block, LDS_BYTES, stream>>>(
        x, mask, h_w0, h_b0, h_w1, h_b1, h_w2, h_b2,
        r_w0, r_b0, r_w1, r_b1, r_w2, r_b2, out);
}
